// DACESR_1365799600944
// MI455X (gfx1250) — compile-verified, matrix operand order checked by reading, not run on hardware
//
#include <hip/hip_runtime.h>

#define DEV_INLINE __device__ __forceinline__

namespace {
constexpr int Bb    = 4;
constexpr int Ll    = 4096;
constexpr int Dd    = 96;
constexpr int DEPTH = 6;
constexpr int DIN   = 192;
constexpr int NST   = 16;
constexpr int RNK   = 6;
constexpr int KC    = 4;
constexpr int Mrows = Bb * Ll;     // 16384
constexpr int NPROJ = 2 * DIN;     // 384
constexpr int XPAD  = 48;          // x_proj rows padded 38 -> 48
constexpr int K2D   = Dd * 9;      // 864 (implicit-GEMM K for 3x3 conv)
}

typedef _Float16 half16  __attribute__((ext_vector_type(16)));
typedef _Float16 half8v  __attribute__((ext_vector_type(8)));
typedef float    floatx8 __attribute__((ext_vector_type(8)));
typedef int      v4i_gcc __attribute__((vector_size(16)));

// ---------------------------------------------------------------------------
// CDNA5 async global->LDS path (ASYNCcnt-tracked).
// ---------------------------------------------------------------------------
#if defined(__HIP_DEVICE_COMPILE__)
#if __has_builtin(__builtin_amdgcn_global_load_async_to_lds_b128)
#define HAVE_ASYNC_LDS 1
#endif
#endif

DEV_INLINE void async_copy16(const _Float16* g, _Float16* l) {
#if HAVE_ASYNC_LDS
  __builtin_amdgcn_global_load_async_to_lds_b128(
      (__attribute__((address_space(1))) v4i_gcc*)g,
      (__attribute__((address_space(3))) v4i_gcc*)l, 0, 0);
#else
  *reinterpret_cast<int4*>(l) = *reinterpret_cast<const int4*>(g);
#endif
}

DEV_INLINE void async_join() {
#if HAVE_ASYNC_LDS
#if __has_builtin(__builtin_amdgcn_s_wait_asynccnt)
  __builtin_amdgcn_s_wait_asynccnt(0);
#else
  asm volatile("s_wait_asynccnt 0x0" ::: "memory");
#endif
#endif
}

// ---------------------------------------------------------------------------
// WMMA fragment loaders. Per-lane fragment = two contiguous 16B runs:
//   k = half*8 + [0..7]  and  k = 16 + half*8 + [0..7]
// -> two b128 loads whenever the row pitch (halves) is a multiple of 8.
// ---------------------------------------------------------------------------
DEV_INLINE half16 pack_frag(half8v lo, half8v hi) {
  half16 f;
#pragma unroll
  for (int i = 0; i < 8; ++i) { f[i] = lo[i]; f[8 + i] = hi[i]; }
  return f;
}

DEV_INLINE half16 load_frag16(const _Float16* __restrict__ p, int ld, int row0,
                              int k0, int lane) {
  const int half = (lane >> 4) & 1;
  const int r    = lane & 15;
  const _Float16* base = p + (long)(row0 + r) * ld + k0 + half * 8;
  half8v lo = *reinterpret_cast<const half8v*>(base);
  half8v hi = *reinterpret_cast<const half8v*>(base + 16);
  return pack_frag(lo, hi);
}

template <int LDB>
DEV_INLINE half16 load_frag_lds(const _Float16* base, int ks, int lane) {
  const int half = (lane >> 4) & 1;
  const int r    = lane & 15;
  const _Float16* rp = base + r * LDB + ks + half * 8;
  half8v lo = *reinterpret_cast<const half8v*>(rp);
  half8v hi = *reinterpret_cast<const half8v*>(rp + 16);
  return pack_frag(lo, hi);
}

// ---------------------------------------------------------------------------
// Staged WMMA GEMM: C(M,N) = A(M,K) * W(N,K)^T (+ optional bias/residual).
// Block: 8 waves, tile 128M x (NT*16)N. B panel staged into LDS per K-chunk
// via async-to-LDS. Per k-step: preload A-frag + NT B-frags, then issue NT
// WMMAs back-to-back.
// ---------------------------------------------------------------------------
template <int NT, int KTOT, int KCHUNK>
__global__ void __launch_bounds__(256) gemm_f16_staged(
    const _Float16* __restrict__ A, const _Float16* __restrict__ W,
    float* __restrict__ C, int N,
    const float* __restrict__ bias, const float* __restrict__ res) {
  constexpr int LDB = KCHUNK + 8;      // keep 16B alignment of frag runs
  __shared__ _Float16 ldsB[NT * 16][LDB];
  const int lane = threadIdx.x & 31;
  const int wave = threadIdx.x >> 5;
  const int row0 = blockIdx.x * 128 + wave * 16;
  const int n0   = blockIdx.y * (NT * 16);
  floatx8 acc[NT] = {};

#pragma unroll
  for (int kc = 0; kc < KTOT; kc += KCHUNK) {
    if (kc) __syncthreads();
    constexpr int CPR  = KCHUNK / 8;         // 16B chunks per row
    constexpr int TOTC = NT * 16 * CPR;
    for (int c = threadIdx.x; c < TOTC; c += 256) {
      const int rrow = c / CPR;
      const int cc   = c - rrow * CPR;
      async_copy16(W + (long)(n0 + rrow) * KTOT + kc + cc * 8,
                   &ldsB[rrow][cc * 8]);
    }
    async_join();
    __syncthreads();
#pragma unroll
    for (int ks = 0; ks < KCHUNK; ks += 32) {
      half16 af = load_frag16(A, KTOT, row0, kc + ks, lane);
      half16 bf[NT];
#pragma unroll
      for (int t = 0; t < NT; ++t)
        bf[t] = load_frag_lds<LDB>(&ldsB[t * 16][0], ks, lane);
#pragma unroll
      for (int t = 0; t < NT; ++t)
        acc[t] = __builtin_amdgcn_wmma_f32_16x16x32_f16(
            false, af, false, bf[t], (short)0, acc[t], false, false);
    }
  }
  const int nn = lane & 15;
  const int mb = row0 + ((lane >> 4) << 3);
  if (bias != nullptr) {  // wave-uniform
#pragma unroll
    for (int t = 0; t < NT; ++t) {
      const int n = n0 + t * 16 + nn;
      const float bv = bias[n];
#pragma unroll
      for (int r = 0; r < 8; ++r) {
        const long mm = mb + r;
        C[mm * N + n] = acc[t][r] + bv + res[mm * N + n];
      }
    }
  } else {
#pragma unroll
    for (int t = 0; t < NT; ++t) {
      const int n = n0 + t * 16 + nn;
#pragma unroll
      for (int r = 0; r < 8; ++r) C[(long)(mb + r) * N + n] = acc[t][r];
    }
  }
}

// ---------------------------------------------------------------------------
// Branchless im2col for the 3x3 conv: A2(M, 864) f16, k = c*9 + ky*3 + kx.
// Clamped-address load + select (no divergent loads); coalesced stores.
// ---------------------------------------------------------------------------
__global__ void __launch_bounds__(256) im2col_3x3(
    const _Float16* __restrict__ img, _Float16* __restrict__ A2) {
  const long i = (long)blockIdx.x * blockDim.x + threadIdx.x;
  if (i >= (long)Mrows * K2D) return;
  const int  k   = (int)(i % K2D);
  const long m   = i / K2D;
  const int  b   = (int)(m >> 12);
  const int  pix = (int)(m & 4095);
  const int  hh  = pix >> 6, ww = pix & 63;
  const int  c   = k / 9;
  const int  t   = k - c * 9;
  const int  ky  = t / 3;
  const int  kx  = t - ky * 3;
  const int  ih  = hh + ky - 1;
  const int  iw  = ww + kx - 1;
  const bool ok  = ((unsigned)ih < 64u) & ((unsigned)iw < 64u);
  const int  ihc = ih < 0 ? 0 : (ih > 63 ? 63 : ih);
  const int  iwc = iw < 0 ? 0 : (iw > 63 ? 63 : iw);
  const _Float16 v = img[(((long)b << 12) + (ihc << 6) + iwc) * Dd + c];
  A2[i] = ok ? v : (_Float16)0.f;
}

// ---------------------------------------------------------------------------
// LayerNorm over D=96, one wave per row (3 elems/lane), output f16.
// ---------------------------------------------------------------------------
__global__ void __launch_bounds__(256) layernorm_to_f16(
    const float* __restrict__ src, const float* __restrict__ w,
    const float* __restrict__ b, _Float16* __restrict__ dst) {
  const int row  = blockIdx.x * 8 + (threadIdx.x >> 5);
  const int lane = threadIdx.x & 31;
  const float* hr = src + (long)row * Dd;
  float v0 = hr[lane], v1 = hr[lane + 32], v2 = hr[lane + 64];
  float s = v0 + v1 + v2;
#pragma unroll
  for (int off = 16; off > 0; off >>= 1) s += __shfl_xor(s, off, 32);
  const float mu = s * (1.f / 96.f);
  const float d0 = v0 - mu, d1 = v1 - mu, d2 = v2 - mu;
  float var = d0 * d0 + d1 * d1 + d2 * d2;
#pragma unroll
  for (int off = 16; off > 0; off >>= 1) var += __shfl_xor(var, off, 32);
  const float rs = rsqrtf(var * (1.f / 96.f) + 1e-5f);
  _Float16* out = dst + (long)row * Dd;
  out[lane]      = (_Float16)(d0 * rs * w[lane]      + b[lane]);
  out[lane + 32] = (_Float16)(d1 * rs * w[lane + 32] + b[lane + 32]);
  out[lane + 64] = (_Float16)(d2 * rs * w[lane + 64] + b[lane + 64]);
}

// ---------------------------------------------------------------------------
// Depthwise causal conv1d (KC=4, pad left 3) + SiLU; writes f32 + f16 copies.
// ---------------------------------------------------------------------------
__global__ void __launch_bounds__(256) conv1d_silu(
    const float* __restrict__ xz, const float* __restrict__ cw,
    const float* __restrict__ cb, float* __restrict__ xs,
    _Float16* __restrict__ xs16) {
  const long i = (long)blockIdx.x * blockDim.x + threadIdx.x;
  if (i >= (long)Mrows * DIN) return;
  const int  d  = (int)(i % DIN);
  const long m  = i / DIN;
  const int  l  = (int)(m % Ll);
  const long bL = m - l;
  float acc = cb[d];
#pragma unroll
  for (int j = 0; j < KC; ++j) {
    const int ls = l - (KC - 1) + j;
    if (ls >= 0) acc += cw[d * KC + j] * xz[(bL + ls) * NPROJ + d];
  }
  const float sv = acc / (1.f + __expf(-acc));
  xs[i]   = sv;
  xs16[i] = (_Float16)sv;
}

// ---------------------------------------------------------------------------
// dt = softplus(x_dbl[:, :6] @ dtw^T + dtb)  (K=6: scalar VALU)
// ---------------------------------------------------------------------------
__global__ void __launch_bounds__(256) dt_softplus(
    const float* __restrict__ xdbl, const float* __restrict__ dw,
    const float* __restrict__ db, float* __restrict__ dt) {
  const long i = (long)blockIdx.x * blockDim.x + threadIdx.x;
  if (i >= (long)Mrows * DIN) return;
  const int  d = (int)(i % DIN);
  const long m = i / DIN;
  float acc = db[d];
#pragma unroll
  for (int r = 0; r < RNK; ++r) acc += xdbl[m * XPAD + r] * dw[d * RNK + r];
  dt[i] = (acc > 20.f) ? acc : log1pf(__expf(acc));
}

// ---------------------------------------------------------------------------
// Selective scan: 16 lanes per (b,d) group, one SSM state n per lane.
// ---------------------------------------------------------------------------
__global__ void __launch_bounds__(256) scan_kernel(
    const float* __restrict__ u, const float* __restrict__ dt,
    const float* __restrict__ xdbl, const float* __restrict__ A_log,
    const float* __restrict__ Dskip, float* __restrict__ y) {
  const int tid = blockIdx.x * blockDim.x + threadIdx.x;
  const int g   = tid >> 4;
  const int n   = tid & 15;
  if (g >= Bb * DIN) return;
  const int d = g % DIN;
  const int b = g / DIN;
  const float Ac = -__expf(A_log[d * NST + n]);
  const float Dv = Dskip[d];
  float hs = 0.f;
  const long base = (long)b * Ll;
  for (int l = 0; l < Ll; ++l) {
    const long m = base + l;
    if (l + 8 < Ll) {
      __builtin_prefetch(&dt[(m + 8) * DIN + d], 0, 0);
      __builtin_prefetch(&xdbl[(m + 8) * XPAD + RNK + n], 0, 0);
    }
    const float dtv = dt[m * DIN + d];
    const float uv  = u[m * DIN + d];
    const float Bv  = xdbl[m * XPAD + RNK + n];
    const float Cv  = xdbl[m * XPAD + RNK + NST + n];
    hs = hs * __expf(dtv * Ac) + dtv * Bv * uv;
    float yp = hs * Cv;
    yp += __shfl_xor(yp, 1, 32);
    yp += __shfl_xor(yp, 2, 32);
    yp += __shfl_xor(yp, 4, 32);
    yp += __shfl_xor(yp, 8, 32);
    if (n == 0) y[m * DIN + d] = yp + uv * Dv;
  }
}

// ---------------------------------------------------------------------------
// y <- y * silu(z), output f16 for the out_proj WMMA.
// ---------------------------------------------------------------------------
__global__ void __launch_bounds__(256) gate_silu(
    const float* __restrict__ y, const float* __restrict__ xz,
    _Float16* __restrict__ y16) {
  const long i = (long)blockIdx.x * blockDim.x + threadIdx.x;
  if (i >= (long)Mrows * DIN) return;
  const int  d = (int)(i % DIN);
  const long m = i / DIN;
  const float z = xz[m * NPROJ + DIN + d];
  y16[i] = (_Float16)(y[i] * (z / (1.f + __expf(-z))));
}

// ---------------------------------------------------------------------------
// Weight / activation conversions
// ---------------------------------------------------------------------------
__global__ void __launch_bounds__(256) cvt_f32_f16(
    const float* __restrict__ src, _Float16* __restrict__ dst, int n) {
  const int i = blockIdx.x * blockDim.x + threadIdx.x;
  if (i < n) dst[i] = (_Float16)src[i];
}

__global__ void __launch_bounds__(256) cvt_xproj_pad(
    const float* __restrict__ src, _Float16* __restrict__ dst) {
  const int i = blockIdx.x * blockDim.x + threadIdx.x;
  constexpr int tot = DEPTH * XPAD * DIN;
  if (i >= tot) return;
  const int dly = i / (XPAD * DIN);
  const int rem = i - dly * (XPAD * DIN);
  const int row = rem / DIN;
  const int col = rem - row * DIN;
  float v = 0.f;
  if (row < RNK + 2 * NST) v = src[(dly * (RNK + 2 * NST) + row) * DIN + col];
  dst[i] = (_Float16)v;
}

// ---------------------------------------------------------------------------
extern "C" void kernel_launch(void* const* d_in, const int* in_sizes, int n_in,
                              void* d_out, int out_size, void* d_ws,
                              size_t ws_size, hipStream_t stream) {
  (void)in_sizes; (void)n_in; (void)out_size; (void)ws_size;
  const float* x      = (const float*)d_in[0];
  const float* norm_w = (const float*)d_in[3];
  const float* norm_b = (const float*)d_in[4];
  const float* in_w   = (const float*)d_in[5];
  const float* c1_w   = (const float*)d_in[6];
  const float* c1_b   = (const float*)d_in[7];
  const float* xp_w   = (const float*)d_in[8];
  const float* dtp_w  = (const float*)d_in[9];
  const float* dtp_b  = (const float*)d_in[10];
  const float* A_log  = (const float*)d_in[11];
  const float* D_skip = (const float*)d_in[12];
  const float* out_w  = (const float*)d_in[13];
  const float* c2_w   = (const float*)d_in[14];
  const float* c2_b   = (const float*)d_in[15];
  float* outp = (float*)d_out;

  char*  ws = (char*)d_ws;
  size_t o  = 0;
  auto alloc = [&](size_t bytes) -> char* {
    char* p = ws + o;
    o += (bytes + 255) & ~(size_t)255;
    return p;
  };
  float*    hbuf  = (float*)alloc((size_t)Mrows * Dd * 4);
  _Float16* hn16  = (_Float16*)alloc((size_t)Mrows * Dd * 2);
  float*    xz    = (float*)alloc((size_t)Mrows * NPROJ * 4);
  float*    xs    = (float*)alloc((size_t)Mrows * DIN * 4);
  _Float16* xs16  = (_Float16*)alloc((size_t)Mrows * DIN * 2);
  float*    xdbl  = (float*)alloc((size_t)Mrows * XPAD * 4);
  float*    dtb   = (float*)alloc((size_t)Mrows * DIN * 4);
  float*    ybuf  = (float*)alloc((size_t)Mrows * DIN * 4);
  _Float16* y16   = (_Float16*)alloc((size_t)Mrows * DIN * 2);
  _Float16* h16   = (_Float16*)alloc((size_t)Mrows * Dd * 2);
  _Float16* A2col = (_Float16*)alloc((size_t)Mrows * K2D * 2);
  _Float16* Win   = (_Float16*)alloc((size_t)DEPTH * NPROJ * Dd * 2);
  _Float16* Wxp   = (_Float16*)alloc((size_t)DEPTH * XPAD * DIN * 2);
  _Float16* Wout  = (_Float16*)alloc((size_t)DEPTH * Dd * DIN * 2);
  _Float16* Wc2   = (_Float16*)alloc((size_t)Dd * K2D * 2);

  auto blk = [](long n) { return (unsigned)((n + 255) / 256); };

  // Weight conversion (deterministic, every call).
  cvt_f32_f16<<<blk(DEPTH * NPROJ * Dd), 256, 0, stream>>>(in_w, Win,
                                                           DEPTH * NPROJ * Dd);
  cvt_xproj_pad<<<blk(DEPTH * XPAD * DIN), 256, 0, stream>>>(xp_w, Wxp);
  cvt_f32_f16<<<blk(DEPTH * Dd * DIN), 256, 0, stream>>>(out_w, Wout,
                                                         DEPTH * Dd * DIN);
  cvt_f32_f16<<<blk(Dd * K2D), 256, 0, stream>>>(c2_w, Wc2, Dd * K2D);

  const long nBD    = (long)Mrows * DIN;
  const unsigned MB = Mrows / 128;  // 128 M-blocks
  for (int i = 0; i < DEPTH; ++i) {
    const float* src = (i == 0) ? x : hbuf;
    layernorm_to_f16<<<Mrows / 8, 256, 0, stream>>>(src, norm_w + i * Dd,
                                                    norm_b + i * Dd, hn16);
    // in_proj: (M,96) x (384,96)^T -> (M,384); NT=4 (64 N per block)
    gemm_f16_staged<4, Dd, Dd><<<dim3(MB, NPROJ / 64), 256, 0, stream>>>(
        hn16, Win + (size_t)i * NPROJ * Dd, xz, NPROJ, nullptr, nullptr);
    conv1d_silu<<<blk(nBD), 256, 0, stream>>>(xz, c1_w + i * DIN * KC,
                                              c1_b + i * DIN, xs, xs16);
    // x_proj: (M,192) x (48,192)^T -> (M,48); NT=3 covers all of N
    gemm_f16_staged<3, DIN, DIN><<<dim3(MB, 1), 256, 0, stream>>>(
        xs16, Wxp + (size_t)i * XPAD * DIN, xdbl, XPAD, nullptr, nullptr);
    dt_softplus<<<blk(nBD), 256, 0, stream>>>(xdbl, dtp_w + i * DIN * RNK,
                                              dtp_b + i * DIN, dtb);
    scan_kernel<<<(Bb * DIN * 16) / 256, 256, 0, stream>>>(
        xs, dtb, xdbl, A_log + i * DIN * NST, D_skip + i * DIN, ybuf);
    gate_silu<<<blk(nBD), 256, 0, stream>>>(ybuf, xz, y16);
    // out_proj: (M,192) x (96,192)^T -> (M,96); NT=6 covers all of N
    gemm_f16_staged<6, DIN, DIN><<<dim3(MB, 1), 256, 0, stream>>>(
        y16, Wout + (size_t)i * Dd * DIN, hbuf, Dd, nullptr, nullptr);
  }
  cvt_f32_f16<<<blk((long)Mrows * Dd), 256, 0, stream>>>(hbuf, h16, Mrows * Dd);
  // conv2d = im2col + staged WMMA GEMM with fused bias + residual -> d_out
  im2col_3x3<<<blk((long)Mrows * K2D), 256, 0, stream>>>(h16, A2col);
  gemm_f16_staged<6, K2D, 288><<<dim3(MB, 1), 256, 0, stream>>>(
      A2col, Wc2, outp, Dd, c2_b, x);
}
